// AttnBlock_28948079575223
// MI455X (gfx1250) — matrix operand order repaired and checked by reading, not compiled, not run on hardware
//
#include <hip/hip_runtime.h>

typedef __attribute__((ext_vector_type(16))) _Float16 v16h;
typedef __attribute__((ext_vector_type(8)))  _Float16 v8h;
typedef __attribute__((ext_vector_type(8)))  float    v8f;
typedef __attribute__((ext_vector_type(4)))  float    v4f;

#define NH  8
#define DH  32
#define CCH 256
#define SS  4096
#define BB  2

// ---------------------------------------------------------------------------
// Kernel 0: BatchNorm (eval) + transpose  x[b][c][s] (f32) -> xT[b][s][c] (f16)
// ---------------------------------------------------------------------------
__global__ __launch_bounds__(256) void k_bn_xpose(
    const float* __restrict__ x, const float* __restrict__ mean,
    const float* __restrict__ var, const float* __restrict__ gamma,
    const float* __restrict__ beta, _Float16* __restrict__ xT) {
  __shared__ float tile[32][33];
  int b  = blockIdx.z;
  int s0 = blockIdx.x * 32, c0 = blockIdx.y * 32;
  int tx = threadIdx.x, ty = threadIdx.y;
  const float* xb = x + (size_t)b * CCH * SS;
#pragma unroll
  for (int i = 0; i < 4; ++i) {
    int c = c0 + ty + i * 8;
    tile[ty + i * 8][tx] = xb[(size_t)c * SS + s0 + tx];   // coalesced in s
  }
  __syncthreads();
  int c = c0 + tx;
  float scale = gamma[c] * rsqrtf(var[c] + 1e-5f);
  float shift = beta[c] - mean[c] * scale;
  _Float16* o = xT + (size_t)b * SS * CCH;
#pragma unroll
  for (int i = 0; i < 4; ++i) {
    int s = s0 + ty + i * 8;
    o[(size_t)s * CCH + c] = (_Float16)(tile[tx][ty + i * 8] * scale + shift);
  }
}

// ---------------------------------------------------------------------------
// Kernel 1: QKV GEMM  qkv[o][b][s] = w_qkv[o][:] . xT[b][s][:]
// wave-level 16x64 output tile, K-loop of 32, v_wmma_f32_16x16x32_f16.
// Epilogue routes into Q[b,h,s,d] (scaled by d^-0.5 * log2(e)),
// K[b,h,s,d], V[b,h,d,s] (f16).
// ---------------------------------------------------------------------------
__global__ __launch_bounds__(128) void k_qkv(
    const float* __restrict__ wqkv, const _Float16* __restrict__ xT,
    _Float16* __restrict__ Qw, _Float16* __restrict__ Kw,
    _Float16* __restrict__ Vw) {
  int b = blockIdx.z;
  int wave = threadIdx.x >> 5, lane = threadIdx.x & 31;
  int half16 = lane >> 4, l16 = lane & 15;
  int o_tile = blockIdx.x * 64 + wave * 16;
  int s_tile = blockIdx.y * 64;
  const _Float16* xTb = xT + (size_t)b * SS * CCH;

  v8f acc[4] = {};
  for (int k0 = 0; k0 < CCH; k0 += 32) {
    // A operand: row o_tile+l16, K 0..15 (+16 if lane>=16)
    union { v16h v; v8h h[2]; } A;
    {
      const float* wrow = wqkv + (size_t)(o_tile + l16) * CCH + k0 + half16 * 16;
      v4f f0 = *(const v4f*)(wrow);
      v4f f1 = *(const v4f*)(wrow + 4);
      v4f f2 = *(const v4f*)(wrow + 8);
      v4f f3 = *(const v4f*)(wrow + 12);
      v8h h0, h1;
#pragma unroll
      for (int i = 0; i < 4; ++i) {
        h0[i] = (_Float16)f0[i]; h0[4 + i] = (_Float16)f1[i];
        h1[i] = (_Float16)f2[i]; h1[4 + i] = (_Float16)f3[i];
      }
      A.h[0] = h0; A.h[1] = h1;
    }
#pragma unroll
    for (int j = 0; j < 4; ++j) {
      int s = s_tile + j * 16 + l16;        // B col = lane&15, K half by lane>>4
      v16h Bv = *(const v16h*)(xTb + (size_t)s * CCH + k0 + half16 * 16);
      acc[j] = __builtin_amdgcn_wmma_f32_16x16x32_f16(
          false, A.v, false, Bv, (short)0, acc[j], false, false);
    }
  }
  // d^-0.5 * log2(e): puts QK^T logits directly in exp2 domain for softmax
  const float qscale = 0.1767766952966369f * 1.4426950408889634f;
#pragma unroll
  for (int j = 0; j < 4; ++j) {
#pragma unroll
    for (int r = 0; r < 8; ++r) {
      int o = o_tile + r + half16 * 8;        // C layout: M = r (+8 hi half)
      int s = s_tile + j * 16 + l16;
      float val = acc[j][r];
      if (o < 256) {
        int head = o >> 5, dd = o & 31;
        Qw[((size_t)(b * NH + head) * SS + s) * DH + dd] = (_Float16)(val * qscale);
      } else if (o < 512) {
        int oc = o - 256, head = oc >> 5, dd = oc & 31;
        Kw[((size_t)(b * NH + head) * SS + s) * DH + dd] = (_Float16)val;
      } else {
        int oc = o - 512, head = oc >> 5, dd = oc & 31;
        Vw[((size_t)(b * NH + head) * DH + dd) * SS + s] = (_Float16)val;
      }
    }
  }
}

// ---------------------------------------------------------------------------
// Kernel 2: flash attention. One wave per 32-query tile (two A operands
// sharing every K/V B operand). Per 32-key step: 4 QK wmma + online softmax
// in exp2 domain (per-lane deferred denominator, shfl max-reduce only) +
// LDS transpose of P into A layout + 4 P.V wmma. Output stored to the
// torch-faithful scrambled layout attn[b][(h*S+s)*d+dd] (f16).
// ---------------------------------------------------------------------------
__global__ __launch_bounds__(128) void k_attn(
    const _Float16* __restrict__ Qw, const _Float16* __restrict__ Kw,
    const _Float16* __restrict__ Vw, _Float16* __restrict__ attn) {
  __shared__ _Float16 Pbuf[4][2][16][32];
  int wave = threadIdx.x >> 5, lane = threadIdx.x & 31;
  int half16 = lane >> 4, l16 = lane & 15;
  int head = blockIdx.y, b = blockIdx.z;
  int bh = b * NH + head;
  int s_tile = (blockIdx.x * 4 + wave) * 32;
  const _Float16* Qh = Qw + (size_t)bh * SS * DH;
  const _Float16* Kh = Kw + (size_t)bh * SS * DH;
  const _Float16* Vh = Vw + (size_t)bh * DH * SS;

  union { v16h v; v8h h[2]; } Aq[2];
#pragma unroll
  for (int t = 0; t < 2; ++t) {
    const _Float16* qrow = Qh + (size_t)(s_tile + t * 16 + l16) * DH + half16 * 16;
    Aq[t].h[0] = *(const v8h*)(qrow);
    Aq[t].h[1] = *(const v8h*)(qrow + 8);
  }

  v8f O[2][2] = {};
  float rmax[2][8], rsum[2][8];
#pragma unroll
  for (int t = 0; t < 2; ++t)
#pragma unroll
    for (int r = 0; r < 8; ++r) { rmax[t][r] = -1e30f; rsum[t][r] = 0.f; }

  for (int kb = 0; kb < SS; kb += 32) {
    v16h K0 = *(const v16h*)(Kh + (size_t)(kb + l16) * DH + half16 * 16);
    v16h K1 = *(const v16h*)(Kh + (size_t)(kb + 16 + l16) * DH + half16 * 16);
    v8f sc[2][2];
#pragma unroll
    for (int t = 0; t < 2; ++t) {
      v8f z0 = {}, z1 = {};
      sc[t][0] = __builtin_amdgcn_wmma_f32_16x16x32_f16(false, Aq[t].v, false, K0, (short)0, z0, false, false);
      sc[t][1] = __builtin_amdgcn_wmma_f32_16x16x32_f16(false, Aq[t].v, false, K1, (short)0, z1, false, false);
    }

#pragma unroll
    for (int t = 0; t < 2; ++t) {
#pragma unroll
      for (int r = 0; r < 8; ++r) {
        // logits are already in exp2 domain (log2e folded into Q)
        float m = fmaxf(sc[t][0][r], sc[t][1][r]);
#pragma unroll
        for (int off = 8; off >= 1; off >>= 1) m = fmaxf(m, __shfl_xor(m, off, 32));
        float nm = fmaxf(rmax[t][r], m);
        float corr = exp2f(rmax[t][r] - nm);
        float e0 = exp2f(sc[t][0][r] - nm);
        float e1 = exp2f(sc[t][1][r] - nm);
        rsum[t][r] = rsum[t][r] * corr + (e0 + e1);   // per-lane partial sum
        rmax[t][r] = nm;
        O[t][0][r] *= corr; O[t][1][r] *= corr;
        int mrow = r + half16 * 8;
        Pbuf[wave][t][mrow][l16]      = (_Float16)e0;   // C-layout -> LDS
        Pbuf[wave][t][mrow][l16 + 16] = (_Float16)e1;
      }
    }
    asm volatile("s_wait_dscnt 0" ::: "memory");

    v16h V0 = *(const v16h*)(Vh + (size_t)(l16)      * SS + kb + half16 * 16);
    v16h V1 = *(const v16h*)(Vh + (size_t)(l16 + 16) * SS + kb + half16 * 16);
#pragma unroll
    for (int t = 0; t < 2; ++t) {
      union { v16h v; v8h h[2]; } Ap;                  // reload in A layout
      Ap.h[0] = *(const v8h*)(&Pbuf[wave][t][l16][half16 * 16]);
      Ap.h[1] = *(const v8h*)(&Pbuf[wave][t][l16][half16 * 16 + 8]);
      O[t][0] = __builtin_amdgcn_wmma_f32_16x16x32_f16(false, Ap.v, false, V0, (short)0, O[t][0], false, false);
      O[t][1] = __builtin_amdgcn_wmma_f32_16x16x32_f16(false, Ap.v, false, V1, (short)0, O[t][1], false, false);
    }
  }

  _Float16* ab = attn + (size_t)b * SS * CCH;
#pragma unroll
  for (int t = 0; t < 2; ++t) {
#pragma unroll
    for (int r = 0; r < 8; ++r) {
      float sum = rsum[t][r];                // reduce deferred denominator once
#pragma unroll
      for (int off = 8; off >= 1; off >>= 1) sum += __shfl_xor(sum, off, 32);
      float inv = 1.0f / sum;
      int mrow = r + half16 * 8;
      int s = s_tile + t * 16 + mrow;
      ab[((size_t)head * SS + s) * DH + l16]      = (_Float16)(O[t][0][r] * inv);
      ab[((size_t)head * SS + s) * DH + l16 + 16] = (_Float16)(O[t][1][r] * inv);
    }
  }
}

// ---------------------------------------------------------------------------
// Kernel 3: proj GEMM + residual. B-operand read directly from attn (s-major).
// ---------------------------------------------------------------------------
__global__ __launch_bounds__(128) void k_proj(
    const float* __restrict__ wproj, const _Float16* __restrict__ attn,
    const float* __restrict__ x, float* __restrict__ out) {
  int b = blockIdx.z;
  int wave = threadIdx.x >> 5, lane = threadIdx.x & 31;
  int half16 = lane >> 4, l16 = lane & 15;
  int o_tile = blockIdx.x * 64 + wave * 16;
  int s_tile = blockIdx.y * 64;
  const _Float16* ab = attn + (size_t)b * SS * CCH;

  v8f acc[4] = {};
  for (int k0 = 0; k0 < CCH; k0 += 32) {
    union { v16h v; v8h h[2]; } A;
    {
      const float* wrow = wproj + (size_t)(o_tile + l16) * CCH + k0 + half16 * 16;
      v4f f0 = *(const v4f*)(wrow);
      v4f f1 = *(const v4f*)(wrow + 4);
      v4f f2 = *(const v4f*)(wrow + 8);
      v4f f3 = *(const v4f*)(wrow + 12);
      v8h h0, h1;
#pragma unroll
      for (int i = 0; i < 4; ++i) {
        h0[i] = (_Float16)f0[i]; h0[4 + i] = (_Float16)f1[i];
        h1[i] = (_Float16)f2[i]; h1[4 + i] = (_Float16)f3[i];
      }
      A.h[0] = h0; A.h[1] = h1;
    }
#pragma unroll
    for (int j = 0; j < 4; ++j) {
      int s = s_tile + j * 16 + l16;
      v16h Bv = *(const v16h*)(ab + (size_t)s * CCH + k0 + half16 * 16);
      acc[j] = __builtin_amdgcn_wmma_f32_16x16x32_f16(
          false, A.v, false, Bv, (short)0, acc[j], false, false);
    }
  }
  const float* xb = x + (size_t)b * CCH * SS;
  float* ob = out + (size_t)b * CCH * SS;
#pragma unroll
  for (int j = 0; j < 4; ++j)
#pragma unroll
    for (int r = 0; r < 8; ++r) {
      int o = o_tile + r + half16 * 8;
      int s = s_tile + j * 16 + l16;
      size_t idx = (size_t)o * SS + s;
      ob[idx] = xb[idx] + acc[j][r];
    }
}

// ---------------------------------------------------------------------------
extern "C" void kernel_launch(void* const* d_in, const int* in_sizes, int n_in,
                              void* d_out, int out_size, void* d_ws, size_t ws_size,
                              hipStream_t stream) {
  const float* x     = (const float*)d_in[0];
  const float* wqkv  = (const float*)d_in[1];
  const float* wproj = (const float*)d_in[2];
  const float* gamma = (const float*)d_in[3];
  const float* beta  = (const float*)d_in[4];
  const float* mean  = (const float*)d_in[5];
  const float* var   = (const float*)d_in[6];

  const size_t N = (size_t)BB * SS * CCH;   // 2,097,152 elements per tensor
  _Float16* xT   = (_Float16*)d_ws;
  _Float16* Qw   = xT + N;
  _Float16* Kw   = Qw + N;
  _Float16* Vw   = Kw + N;
  _Float16* attn = Vw + N;                  // total 20 MB of f16 workspace

  k_bn_xpose<<<dim3(SS / 32, CCH / 32, BB), dim3(32, 8), 0, stream>>>(
      x, mean, var, gamma, beta, xT);
  k_qkv<<<dim3(768 / 64, SS / 64, BB), 128, 0, stream>>>(wqkv, xT, Qw, Kw, Vw);
  k_attn<<<dim3(SS / 128, NH, BB), 128, 0, stream>>>(Qw, Kw, Vw, attn);
  k_proj<<<dim3(CCH / 64, SS / 64, BB), 128, 0, stream>>>(
      wproj, attn, x, (float*)d_out);
}
